// CrossParentAttn_39702677684480
// MI455X (gfx1250) — matrix operand order repaired and checked by reading, not compiled, not run on hardware
//
#include <hip/hip_runtime.h>
#include <hip/hip_bf16.h>

typedef __attribute__((ext_vector_type(16))) _Float16 v16h;
typedef __attribute__((ext_vector_type(8)))  _Float16 v8h;
typedef __attribute__((ext_vector_type(8)))  float    v8f;
typedef __attribute__((ext_vector_type(4)))  float    v4f;
typedef __attribute__((ext_vector_type(4)))  unsigned int u32x4;
typedef __attribute__((ext_vector_type(8)))  int      i32x8;
typedef __attribute__((ext_vector_type(4)))  int      i32x4;

#define C_DIM 128
#define HEADS 2
#define DH    64
#define B_    4
#define T_    64
#define NF    384
#define NJ    (NF / 32)
#define ROWS  (B_*T_*NF)   /* 98304 */

#if defined(__has_builtin)
#  if __has_builtin(__builtin_amdgcn_tensor_load_to_lds) && __has_builtin(__builtin_amdgcn_s_wait_tensorcnt)
#    define USE_TDM 1
#  endif
#endif
#ifndef USE_TDM
#  define USE_TDM 0
#endif

// ---------------- WMMA tile helpers (CDNA5 16-bit layouts, wave32) -----------
// A 16x32: lane L (row M=L%16) holds K = h + 16*(L/16) at half h (same order as B).
// B 32x16: lane L (col N=L%16) holds K = h + 16*(L/16) at half h.
// C/D f32 16x16: vgpr r, lane L -> M = r + 8*(L/16), N = L%16.

__device__ __forceinline__ v8f wmma32(v16h a, v16h b, v8f c) {
  return __builtin_amdgcn_wmma_f32_16x16x32_f16(false, a, false, b, (short)0, c, false, false);
}

__device__ __forceinline__ v16h ldA_f16g(const _Float16* __restrict__ X, int row0, int k0, int lane) {
  const _Float16* p = X + (size_t)(row0 + (lane & 15)) * C_DIM + k0 + ((lane >> 4) << 4);
  v8h lo = *(const v8h*)p;
  v8h hi = *(const v8h*)(p + 8);
  v16h a;
#pragma unroll
  for (int i = 0; i < 8; ++i) { a[i] = lo[i]; a[8 + i] = hi[i]; }
  return a;
}

__device__ __forceinline__ v16h ldA_lds(const _Float16* base, int ld, int row, int k0, int lane) {
  const _Float16* p = base + (row + (lane & 15)) * ld + k0 + ((lane >> 4) << 4);
  v8h lo = *(const v8h*)p;
  v8h hi = *(const v8h*)(p + 8);
  v16h a;
#pragma unroll
  for (int i = 0; i < 8; ++i) { a[i] = lo[i]; a[8 + i] = hi[i]; }
  return a;
}

__device__ __forceinline__ v16h ldB_f16(const _Float16* __restrict__ X, int n0, int k0, int lane) {
  const _Float16* p = X + (size_t)(n0 + (lane & 15)) * C_DIM + k0 + ((lane >> 4) << 4);
  return *(const v16h*)p;
}

// B-tile from LDS tile with row stride 64 halves (contiguous 16-half read)
__device__ __forceinline__ v16h ldB_lds64(const _Float16* base, int n0, int k0, int lane) {
  const _Float16* p = base + (n0 + (lane & 15)) * 64 + k0 + ((lane >> 4) << 4);
  v8h lo = *(const v8h*)p;
  v8h hi = *(const v8h*)(p + 8);
  v16h b;
#pragma unroll
  for (int i = 0; i < 8; ++i) { b[i] = lo[i]; b[8 + i] = hi[i]; }
  return b;
}

#if USE_TDM
// 2D TDM: load a 32-row x 64-halfword tile (global row stride 128 halves) to LDS
__device__ __forceinline__ void tdm_load_tile(const _Float16* gsrc, void* ldst) {
  const unsigned long long ga = (unsigned long long)(size_t)gsrc;
  u32x4 g0;
  g0[0] = 1u;                                                 // count=1, user mode
  g0[1] = (unsigned)(size_t)ldst;                             // lds_addr (low 32 = LDS offset)
  g0[2] = (unsigned)ga;                                       // global_addr[31:0]
  g0[3] = ((unsigned)(ga >> 32) & 0x01FFFFFFu) | (2u << 30);  // addr[56:32] | type=2
  i32x8 g1 = {};
  g1[0] = (int)(1u << 16);                 // data_size = 1 (2 bytes)
  g1[1] = (int)(64u << 16);                // tensor_dim0 = 64         (bits 79:48)
  g1[2] = (int)(32u << 16);                // tensor_dim1 = 32         (bits 111:80)
  g1[3] = (int)(64u << 16);                // tile_dim0   = 64         (bits 127:112)
  g1[4] = (int)(32u);                      // tile_dim1   = 32         (bits 143:128)
  g1[5] = (int)(128u);                     // tensor_dim0_stride = 128 (bits 207:160)
  g1[6] = (int)((4096u & 0xFFFFu) << 16);  // tensor_dim1_stride = 4096 (bits 255:208)
  g1[7] = 0;
  i32x4 z4 = {};
#if __clang_major__ >= 23
  i32x8 z8 = {};
  __builtin_amdgcn_tensor_load_to_lds(g0, g1, z4, z4, z8, 0);
#else
  __builtin_amdgcn_tensor_load_to_lds(g0, g1, z4, z4, 0);
#endif
}
#endif

// ---------------- Weight cast: 128x128 f32 -> f16 ----------------------------
__global__ __launch_bounds__(256) void wcast_kernel(const float* __restrict__ X,
                                                    _Float16* __restrict__ Y) {
  const int i = blockIdx.x * 256 + threadIdx.x;   // grid 64 * 256 = 16384
  Y[i] = (_Float16)X[i];
}

// ---------------- Y(f16) = X(f32) @ Wh^T + bias ------------------------------
__global__ __launch_bounds__(256) void proj_kernel(const float* __restrict__ X,
                                                   const _Float16* __restrict__ Wh,
                                                   const float* __restrict__ bias,
                                                   _Float16* __restrict__ Y) {
  __shared__ _Float16 Alds[64][C_DIM];            // 16 KB
  const int tid  = threadIdx.x;
  const int lane = tid & 31;
  const int wave = tid >> 5;
  const int row0 = blockIdx.x * 64;
  {
    const int r  = tid >> 2;
    const int cb = (tid & 3) * 32;
    const float* xp = X + (size_t)(row0 + r) * C_DIM + cb;
#pragma unroll
    for (int x = 0; x < 4; ++x) {
      v4f f0 = *(const v4f*)(xp + x * 8);
      v4f f1 = *(const v4f*)(xp + x * 8 + 4);
      v8h hbuf;
#pragma unroll
      for (int i = 0; i < 4; ++i) { hbuf[i] = (_Float16)f0[i]; hbuf[4 + i] = (_Float16)f1[i]; }
      *(v8h*)&Alds[r][cb + x * 8] = hbuf;
    }
  }
  __syncthreads();

  const int n0 = wave * 16;
  v8f acc[4] = {};
#pragma unroll
  for (int k0 = 0; k0 < C_DIM; k0 += 32) {
    v16h b = ldB_f16(Wh, n0, k0, lane);
#pragma unroll
    for (int rt = 0; rt < 4; ++rt) {
      v16h a = ldA_lds(&Alds[0][0], C_DIM, rt * 16, k0, lane);
      acc[rt] = wmma32(a, b, acc[rt]);
    }
  }
  const float bv = bias[n0 + (lane & 15)];
  const int m0 = (lane >> 4) << 3;
  const int n  = n0 + (lane & 15);
#pragma unroll
  for (int rt = 0; rt < 4; ++rt)
#pragma unroll
    for (int r = 0; r < 8; ++r)
      Y[(size_t)(row0 + rt * 16 + m0 + r) * C_DIM + n] = (_Float16)(acc[rt][r] + bv);
}

// ---------------- out(f32) = O(f16) @ Woh^T + bo -----------------------------
__global__ __launch_bounds__(256) void oproj_kernel(const _Float16* __restrict__ X,
                                                    const _Float16* __restrict__ Wh,
                                                    const float* __restrict__ bias,
                                                    float* __restrict__ Y) {
  __shared__ _Float16 Alds[64][C_DIM];
  const int tid  = threadIdx.x;
  const int lane = tid & 31;
  const int wave = tid >> 5;
  const int row0 = blockIdx.x * 64;
  {
    const int r  = tid >> 2;
    const int cb = (tid & 3) * 32;
    const _Float16* xp = X + (size_t)(row0 + r) * C_DIM + cb;
#pragma unroll
    for (int x = 0; x < 4; ++x)
      *(v8h*)&Alds[r][cb + x * 8] = *(const v8h*)(xp + x * 8);
  }
  __syncthreads();

  const int n0 = wave * 16;
  v8f acc[4] = {};
#pragma unroll
  for (int k0 = 0; k0 < C_DIM; k0 += 32) {
    v16h b = ldB_f16(Wh, n0, k0, lane);
#pragma unroll
    for (int rt = 0; rt < 4; ++rt) {
      v16h a = ldA_lds(&Alds[0][0], C_DIM, rt * 16, k0, lane);
      acc[rt] = wmma32(a, b, acc[rt]);
    }
  }
  const float bv = bias[n0 + (lane & 15)];
  const int m0 = (lane >> 4) << 3;
  const int n  = n0 + (lane & 15);
#pragma unroll
  for (int rt = 0; rt < 4; ++rt)
#pragma unroll
    for (int r = 0; r < 8; ++r)
      Y[(size_t)(row0 + rt * 16 + m0 + r) * C_DIM + n] = acc[rt][r] + bv;
}

// ---------------- flash attention --------------------------------------------
// grid = (NF/64, B*T*H), block = 128 (4 waves share K/V tiles).
// K tile: TDM double-buffered into LDS (issue j+1, wait, compute j).
// V tile: cooperative transposed staging -> contiguous B-layout LDS reads.
__global__ __launch_bounds__(128) void attn_kernel(const _Float16* __restrict__ Qh,
                                                   const _Float16* __restrict__ Kh,
                                                   const _Float16* __restrict__ Vh,
                                                   _Float16* __restrict__ Oh) {
#if USE_TDM
  __shared__ _Float16 Kbuf[2][32][64];     // 8 KB TDM double buffer
#endif
  __shared__ _Float16 Vt[64][48];          // transposed V tile (6 KB, padded)
  __shared__ _Float16 Plds[4][16][32];     // per-wave P staging (4 KB)

  const int tid  = threadIdx.x;
  const int lane = tid & 31;
  const int wave = tid >> 5;
  const int qt   = blockIdx.x * 4 + wave;  // 0..23
  const int bth  = blockIdx.y;             // 0..511
  const int hh   = bth & (HEADS - 1);
  const int t    = (bth >> 1) & (T_ - 1);
  const int b    = bth >> 7;
  const int base = (b * T_ + t) * NF;
  const int hc   = hh * DH;

  const v16h qa0 = ldA_f16g(Qh, base + qt * 16, hc + 0,  lane);
  const v16h qa1 = ldA_f16g(Qh, base + qt * 16, hc + 32, lane);

  float mi[8], li[8];
  v8f oacc[4];
#pragma unroll
  for (int r = 0; r < 8; ++r) { mi[r] = -1e30f; li[r] = 0.f; }
#pragma unroll
  for (int d = 0; d < 4; ++d) oacc[d] = (v8f){};

  const int g = lane >> 4, c16 = lane & 15;

#if USE_TDM
  if (wave == 0)
    tdm_load_tile(Kh + (size_t)base * C_DIM + hc, &Kbuf[0][0][0]);
#endif

  for (int j = 0; j < NJ; ++j) {
    __builtin_prefetch(Vh + (size_t)(base + j * 32) * C_DIM + hc, 0, 1);

    // ---- S = Q K^T (16 x 32) ----
    v8f s0 = {}, s1 = {};
#if USE_TDM
    if (wave == 0) {
      if (j + 1 < NJ) {
        tdm_load_tile(Kh + (size_t)(base + (j + 1) * 32) * C_DIM + hc,
                      &Kbuf[(j + 1) & 1][0][0]);
        __builtin_amdgcn_s_wait_tensorcnt(1);
      } else {
        __builtin_amdgcn_s_wait_tensorcnt(0);
      }
    }
    __syncthreads();                       // K tile j visible to all waves
    {
      const _Float16* Kt = &Kbuf[j & 1][0][0];
      s0 = wmma32(qa0, ldB_lds64(Kt,  0, 0,  lane), s0);
      s0 = wmma32(qa1, ldB_lds64(Kt,  0, 32, lane), s0);
      s1 = wmma32(qa0, ldB_lds64(Kt, 16, 0,  lane), s1);
      s1 = wmma32(qa1, ldB_lds64(Kt, 16, 32, lane), s1);
    }
#else
    s0 = wmma32(qa0, ldB_f16(Kh, base + j * 32 +  0, hc + 0,  lane), s0);
    s0 = wmma32(qa1, ldB_f16(Kh, base + j * 32 +  0, hc + 32, lane), s0);
    s1 = wmma32(qa0, ldB_f16(Kh, base + j * 32 + 16, hc + 0,  lane), s1);
    s1 = wmma32(qa1, ldB_f16(Kh, base + j * 32 + 16, hc + 32, lane), s1);
#endif

    // ---- online softmax over the 32 new keys ----
    float p0[8], p1[8], alpha[8];
#pragma unroll
    for (int r = 0; r < 8; ++r) {
      float t0 = s0[r] * 0.125f;
      float t1 = s1[r] * 0.125f;
      float mx = fmaxf(t0, t1);
#pragma unroll
      for (int off = 1; off < 16; off <<= 1) mx = fmaxf(mx, __shfl_xor(mx, off, 32));
      float mnew = fmaxf(mi[r], mx);
      alpha[r] = __expf(mi[r] - mnew);
      p0[r] = __expf(t0 - mnew);
      p1[r] = __expf(t1 - mnew);
      float rs = p0[r] + p1[r];
#pragma unroll
      for (int off = 1; off < 16; off <<= 1) rs += __shfl_xor(rs, off, 32);
      li[r] = li[r] * alpha[r] + rs;
      mi[r] = mnew;
    }

    __syncthreads();   // previous iteration's Vt/Plds reads are done

    // ---- stage P (per-wave, C-layout) ----
#pragma unroll
    for (int r = 0; r < 8; ++r) {
      Plds[wave][g * 8 + r][c16]      = (_Float16)p0[r];
      Plds[wave][g * 8 + r][16 + c16] = (_Float16)p1[r];
    }

    // ---- stage V tile transposed: Vt[dim][key] ----
    {
      const int r  = tid >> 2;          // key row 0..31
      const int sb = (tid & 3) * 16;    // dim segment
      const _Float16* vp = Vh + (size_t)(base + j * 32 + r) * C_DIM + hc + sb;
      v8h a0 = *(const v8h*)vp;
      v8h a1 = *(const v8h*)(vp + 8);
#pragma unroll
      for (int i = 0; i < 8; ++i) { Vt[sb + i][r] = a0[i]; Vt[sb + 8 + i][r] = a1[i]; }
    }
    __syncthreads();

    // ---- reload P in A-layout ----
    v16h pa;
    {
      const int off = g << 4;
      v8h lo = *(const v8h*)&Plds[wave][c16][off];
      v8h hi = *(const v8h*)&Plds[wave][c16][off + 8];
#pragma unroll
      for (int i = 0; i < 8; ++i) { pa[i] = lo[i]; pa[8 + i] = hi[i]; }
    }

    // ---- O = alpha*O + P @ V (B-tiles now contiguous in Vt) ----
#pragma unroll
    for (int dt = 0; dt < 4; ++dt) {
#pragma unroll
      for (int r = 0; r < 8; ++r) oacc[dt][r] *= alpha[r];
      const _Float16* p = &Vt[dt * 16 + c16][g << 4];
      v8h lo = *(const v8h*)p;
      v8h hi = *(const v8h*)(p + 8);
      v16h vb;
#pragma unroll
      for (int i = 0; i < 8; ++i) { vb[i] = lo[i]; vb[8 + i] = hi[i]; }
      oacc[dt] = wmma32(pa, vb, oacc[dt]);
    }
  }

  // ---- normalize, store O (f16) ----
#pragma unroll
  for (int dt = 0; dt < 4; ++dt)
#pragma unroll
    for (int r = 0; r < 8; ++r) {
      const int m = (g << 3) + r;
      Oh[(size_t)(base + qt * 16 + m) * C_DIM + hc + dt * 16 + c16] =
          (_Float16)(oacc[dt][r] / li[r]);
    }
}

// -----------------------------------------------------------------------------
extern "C" void kernel_launch(void* const* d_in, const int* in_sizes, int n_in,
                              void* d_out, int out_size, void* d_ws, size_t ws_size,
                              hipStream_t stream) {
  const float* q_in  = (const float*)d_in[0];
  const float* kv_in = (const float*)d_in[1];
  const float* Wq = (const float*)d_in[2];
  const float* bq = (const float*)d_in[3];
  const float* Wk = (const float*)d_in[4];
  const float* bk = (const float*)d_in[5];
  const float* Wv = (const float*)d_in[6];
  const float* bv = (const float*)d_in[7];
  const float* Wo = (const float*)d_in[8];
  const float* bo = (const float*)d_in[9];
  float* out = (float*)d_out;

  _Float16* Qh  = (_Float16*)d_ws;
  _Float16* Kh  = Qh + (size_t)ROWS * C_DIM;
  _Float16* Vh  = Kh + (size_t)ROWS * C_DIM;
  _Float16* Oh  = Vh + (size_t)ROWS * C_DIM;
  _Float16* Whq = Oh + (size_t)ROWS * C_DIM;
  _Float16* Whk = Whq + C_DIM * C_DIM;
  _Float16* Whv = Whk + C_DIM * C_DIM;
  _Float16* Who = Whv + C_DIM * C_DIM;

  hipLaunchKernelGGL(wcast_kernel, dim3(64), dim3(256), 0, stream, Wq, Whq);
  hipLaunchKernelGGL(wcast_kernel, dim3(64), dim3(256), 0, stream, Wk, Whk);
  hipLaunchKernelGGL(wcast_kernel, dim3(64), dim3(256), 0, stream, Wv, Whv);
  hipLaunchKernelGGL(wcast_kernel, dim3(64), dim3(256), 0, stream, Wo, Who);

  dim3 pg(ROWS / 64), pb(256);
  hipLaunchKernelGGL(proj_kernel, pg, pb, 0, stream, q_in,  Whq, bq, Qh);
  hipLaunchKernelGGL(proj_kernel, pg, pb, 0, stream, kv_in, Whk, bk, Kh);
  hipLaunchKernelGGL(proj_kernel, pg, pb, 0, stream, kv_in, Whv, bv, Vh);
  hipLaunchKernelGGL(attn_kernel, dim3(NF / 64, B_ * T_ * HEADS), dim3(128), 0, stream,
                     Qh, Kh, Vh, Oh);
  hipLaunchKernelGGL(oproj_kernel, pg, pb, 0, stream, Oh, Who, bo, out);
}
